// AIO_DownsampleCouplingBlock_326417514501
// MI455X (gfx1250) — matrix operand order repaired and checked by reading, not compiled, not run on hardware
//
#include <hip/hip_runtime.h>
#include <math.h>

typedef __attribute__((ext_vector_type(16))) _Float16 v16h;
typedef __attribute__((ext_vector_type(8)))  _Float16 v8h;
typedef __attribute__((ext_vector_type(8)))  float    v8f;
typedef __attribute__((ext_vector_type(4)))  unsigned int u32x4;
typedef __attribute__((ext_vector_type(8)))  int      i32x8;
typedef __attribute__((ext_vector_type(4)))  int      i32x4;

// ---------------------------------------------------------------------------
// TDM: async 2-D tile load (rows x cols f16 elements, row stride in elements)
// from global to LDS. Descriptor per CDNA5 ISA 8.3/8.4. Issued once per wave.
// ---------------------------------------------------------------------------
__device__ __forceinline__ void tdm_load_2d_f16(unsigned int lds_off,
                                                const _Float16* gptr,
                                                unsigned int rows,
                                                unsigned int cols,
                                                unsigned int row_len,
                                                unsigned long long row_stride) {
  unsigned long long ga = (unsigned long long)(uintptr_t)gptr;
  u32x4 g0;
  g0.x = 1u;                                                   // count=1
  g0.y = lds_off;                                              // lds_addr
  g0.z = (unsigned int)(ga & 0xFFFFFFFFull);                   // global_addr lo
  g0.w = (unsigned int)((ga >> 32) & 0x01FFFFFFull) | (2u << 30); // addr hi | type=2

  i32x8 g1;
  g1[0] = (int)(1u << 16);                                     // data_size=2B
  g1[1] = (int)((row_len & 0xFFFFu) << 16);                    // tensor_dim0[15:0]
  g1[2] = (int)((row_len >> 16) | ((512u & 0xFFFFu) << 16));   // dim0 hi | tensor_dim1 lo
  g1[3] = (int)((512u >> 16) | (cols << 16));                  // dim1 hi | tile_dim0
  g1[4] = (int)(rows & 0xFFFFu);                               // tile_dim1 (tile_dim2=0)
  g1[5] = (int)(unsigned int)(row_stride & 0xFFFFFFFFull);     // dim0_stride lo
  g1[6] = (int)(unsigned int)((row_stride >> 32) & 0xFFFFull); // dim0_stride hi
  g1[7] = 0;

  i32x4 z4 = {0, 0, 0, 0};
#if __has_builtin(__builtin_amdgcn_tensor_load_to_lds)
#if __clang_major__ >= 23
  i32x8 z8 = {0, 0, 0, 0, 0, 0, 0, 0};
  __builtin_amdgcn_tensor_load_to_lds(g0, g1, z4, z4, z8, 0);
#else
  __builtin_amdgcn_tensor_load_to_lds(g0, g1, z4, z4, 0);
#endif
#else
  asm volatile("tensor_load_to_lds %0, %1, %2, %3"
               :: "s"(g0), "s"(g1), "s"(z4), "s"(z4) : "memory");
#endif
}

// ---------------------------------------------------------------------------
// Unified stride-1 implicit-GEMM conv with WMMA f16 (f32 accumulate).
// Input: padded (32, 192, 34, 34) with zero halo; TW = tap width (2 or 3),
// tap offsets (t/TW - 1, t%TW - 1). Weights: tap-major f16 [TW*TW][384][192].
// Output: (32, 384, 32, 32) f32 NCHW.
//
// Block: 256 threads = 8 wave32 waves. Tile: 128 oc x 64 px (two output rows).
// Each wave: 16 oc x 64 px = 4 WMMA 16x16 tiles, 4 f32 accumulators.
// Double-buffered LDS: TDM A-tile load + B-tile staging for chunk i+1 overlap
// the WMMAs of chunk i; one barrier per chunk. Chunk counts are even.
// ---------------------------------------------------------------------------
template <int TW, typename TIN>
__global__ __launch_bounds__(256)
void conv_wmma(const TIN* __restrict__ inp, const _Float16* __restrict__ wT,
               const float* __restrict__ bias, float* __restrict__ out) {
  constexpr int TAPS = TW * TW;
  constexpr int NCH  = TAPS * 6;            // K chunks of 32 over 192*TAPS
  constexpr int OC = 384, PW = 34;
  const int oc0  = blockIdx.x * 128;
  const int h0   = blockIdx.y * 2;
  const int b    = blockIdx.z;
  const int tid  = threadIdx.x;
  const int lane = tid & 31;                // wave32
  const int wave = tid >> 5;

  __shared__ alignas(64) _Float16 As[2][128][32];   // [buf][oc][k] 2 x 8 KB (TDM dest)
  __shared__ alignas(64) _Float16 Bs[2][64][32];    // [buf][n][k]  2 x 4 KB

  v8f acc0 = {}, acc1 = {}, acc2 = {}, acc3 = {};

  const TIN* __restrict__ inb = inp + (size_t)b * 192 * PW * PW;

  auto stage = [&](int buf, int chunk) {
    const int t  = chunk / 6;
    const int k0 = (chunk % 6) * 32;
    const int dh = t / TW - 1, dw = t % TW - 1;
    // A tile via Tensor Data Mover (one issue, wave 0)
    if (wave == 0) {
      const _Float16* gw = wT + ((size_t)t * OC + oc0) * 192 + k0;
      tdm_load_2d_f16((unsigned int)(uintptr_t)&As[buf][0][0], gw,
                      128u, 32u, 192u, 192ull);
    }
    // B tile, branch-free (input pre-padded): Bs[buf][n][kk], n = r*32 + w
    #pragma unroll
    for (int i = 0; i < 8; ++i) {
      const int idx = i * 256 + tid;
      const int n = idx & 63, kk = idx >> 6;
      const int r = n >> 5, w = n & 31;
      const int yy = h0 + r + dh + 1;
      Bs[buf][n][kk] = (_Float16)inb[((size_t)(k0 + kk) * PW + yy) * PW + (dw + 1) + w];
    }
    // prefetch next chunk's channel rows toward the caches
    const int k0n = (k0 + 32 < 192) ? (k0 + 32) : 0;
    __builtin_prefetch(inb + ((size_t)(k0n + (tid >> 3)) * PW + h0 + 1) * PW, 0, 1);
  };

  auto compute = [&](int buf) {
    // A fragment: lane<16 -> M=lane, K 0..15 ; lane>=16 -> K 16..31
    const int am = wave * 16 + (lane & 15);
    const int ab = (lane >> 4) * 16;
    v8h alo = *(const v8h*)&As[buf][am][ab];
    v8h ahi = *(const v8h*)&As[buf][am][ab + 8];
    v16h af = __builtin_shufflevector(alo, ahi,
        0, 1, 2, 3, 4, 5, 6, 7, 8, 9, 10, 11, 12, 13, 14, 15);
    // B fragments: lane<16 -> N=lane, K 0..15 ; lane>=16 -> K 16..31
    const int koff = (lane >> 4) * 16;
    const int nb   = lane & 15;
    v16h b0 = *(const v16h*)&Bs[buf][nb][koff];
    v16h b1 = *(const v16h*)&Bs[buf][16 + nb][koff];
    v16h b2 = *(const v16h*)&Bs[buf][32 + nb][koff];
    v16h b3 = *(const v16h*)&Bs[buf][48 + nb][koff];
    acc0 = __builtin_amdgcn_wmma_f32_16x16x32_f16(false, af, false, b0, (short)0, acc0, false, false);
    acc1 = __builtin_amdgcn_wmma_f32_16x16x32_f16(false, af, false, b1, (short)0, acc1, false, false);
    acc2 = __builtin_amdgcn_wmma_f32_16x16x32_f16(false, af, false, b2, (short)0, acc2, false, false);
    acc3 = __builtin_amdgcn_wmma_f32_16x16x32_f16(false, af, false, b3, (short)0, acc3, false, false);
  };

  // prologue: fill buffer 0
  stage(0, 0);
  if (wave == 0) __builtin_amdgcn_s_wait_tensorcnt(0);
  __syncthreads();

  // ping-pong main loop (NCH is even: 24 or 54)
  for (int c = 0; c < NCH; c += 2) {
    stage(1, c + 1);                     // overlap with compute of buf 0
    compute(0);
    if (wave == 0) __builtin_amdgcn_s_wait_tensorcnt(0);
    __syncthreads();
    if (c + 2 < NCH) stage(0, c + 2);    // overlap with compute of buf 1
    compute(1);
    if (wave == 0) __builtin_amdgcn_s_wait_tensorcnt(0);
    __syncthreads();
  }

  // store: D layout -> oc = base + 8*(lane>=16) + vgpr, col = lane&15
  const int ocw = oc0 + wave * 16 + (lane >> 4) * 8;
  const int nb  = lane & 15;
  #pragma unroll
  for (int v = 0; v < 8; ++v) {
    const int oc = ocw + v;
    const float bi = bias[oc];
    const size_t o0 = (((size_t)b * OC + oc) * 32 + h0) * 32;
    out[o0 + nb]           = acc0[v] + bi;   // row h0,   cols 0..15
    out[o0 + 16 + nb]      = acc1[v] + bi;   // row h0,   cols 16..31
    out[o0 + 32 + nb]      = acc2[v] + bi;   // row h0+1, cols 0..15
    out[o0 + 48 + nb]      = acc3[v] + bi;   // row h0+1, cols 16..31
  }
}

// ---------------------------------------------------------------------------
// perm_w (384x384 one-hot rows) -> index LUT
// ---------------------------------------------------------------------------
__global__ void perm_idx_kernel(const float* __restrict__ perm_w, int* __restrict__ pidx) {
  const int o = blockIdx.x * blockDim.x + threadIdx.x;
  if (o < 384) {
    int idx = 0;
    for (int c = 0; c < 384; ++c)
      if (perm_w[o * 384 + c] > 0.5f) idx = c;
    pidx[o] = idx;
  }
}

// ---------------------------------------------------------------------------
// w_lo (384,192,3,3) OIHW f32 -> tap-major f16 [9][384][192]
// ---------------------------------------------------------------------------
__global__ __launch_bounds__(256)
void wlo_transpose_kernel(const float* __restrict__ w_lo, _Float16* __restrict__ wT) {
  const int idx = blockIdx.x * blockDim.x + threadIdx.x;
  if (idx >= 9 * 384 * 192) return;
  const int ci = idx % 192;
  const int o  = (idx / 192) % 384;
  const int pos = idx / (192 * 384);
  wT[idx] = (_Float16)w_lo[((size_t)o * 192 + ci) * 9 + pos];
}

// ---------------------------------------------------------------------------
// w_hi (384,48,3,3) stride-2 conv -> equivalent 2x2 conv over down(x1):
// f16 [4][384][192], tap t = (dh+1)*2 + (dw+1), dh,dw in {-1,0}.
// ---------------------------------------------------------------------------
__global__ __launch_bounds__(256)
void whi_remap_kernel(const float* __restrict__ w_hi, _Float16* __restrict__ wT) {
  const int idx = blockIdx.x * blockDim.x + threadIdx.x;
  if (idx >= 4 * 384 * 192) return;
  const int ci = idx % 192;
  const int o  = (idx / 192) % 384;
  const int t  = idx / (192 * 384);
  const int dhp = t >> 1, dwp = t & 1;              // 0 -> offset -1, 1 -> offset 0
  const int c0 = ci >> 2, hs = (ci >> 1) & 1, ws = ci & 1;
  const int kh = (dhp == 0) ? (hs == 1 ? 0 : -1) : (hs == 0 ? 1 : 2);
  const int kw = (dwp == 0) ? (ws == 1 ? 0 : -1) : (ws == 0 ? 1 : 2);
  float v = 0.0f;
  if (kh >= 0 && kw >= 0)
    v = w_hi[((size_t)o * 48 + c0) * 9 + kh * 3 + kw];
  wT[idx] = (_Float16)v;
}

// ---------------------------------------------------------------------------
// down(x[:, :48]) -> zero-halo-padded f16 (32,192,34,34)
// ---------------------------------------------------------------------------
__global__ __launch_bounds__(256)
void down_pad_kernel(const float* __restrict__ x, _Float16* __restrict__ d1p) {
  const int idx = blockIdx.x * blockDim.x + threadIdx.x;
  if (idx >= 32 * 192 * 34 * 34) return;
  const int xx = idx % 34;
  const int yy = (idx / 34) % 34;
  const int c  = (idx / (34 * 34)) % 192;
  const int b  = idx / (34 * 34 * 192);
  float v = 0.0f;
  if (xx >= 1 && xx <= 32 && yy >= 1 && yy <= 32) {
    const int h = yy - 1, w = xx - 1;
    v = x[(((size_t)b * 96 + (c >> 2)) * 64 + 2 * h + ((c >> 1) & 1)) * 64 +
          2 * w + (c & 1)];
  }
  d1p[idx] = (_Float16)v;
}

// ---------------------------------------------------------------------------
// y2 = down(x[:,48:]) * exp(2 tanh(0.2 a1_lo)) + a1_hi, written as
// zero-halo-padded f32 (32,192,34,34)
// ---------------------------------------------------------------------------
__global__ __launch_bounds__(256)
void affine_y2p_kernel(const float* __restrict__ x, const float* __restrict__ a1,
                       float* __restrict__ y2p) {
  const int idx = blockIdx.x * blockDim.x + threadIdx.x;
  if (idx >= 32 * 192 * 34 * 34) return;
  const int xx = idx % 34;
  const int yy = (idx / 34) % 34;
  const int c  = (idx / (34 * 34)) % 192;
  const int b  = idx / (34 * 34 * 192);
  float v = 0.0f;
  if (xx >= 1 && xx <= 32 && yy >= 1 && yy <= 32) {
    const int h = yy - 1, w = xx - 1;
    const float dn = x[(((size_t)b * 96 + 48 + (c >> 2)) * 64 + 2 * h + ((c >> 1) & 1)) * 64 +
                       2 * w + (c & 1)];
    const size_t ai = (((size_t)b * 384 + c) * 32 + h) * 32 + w;
    v = dn * expf(2.0f * tanhf(0.2f * a1[ai])) + a1[ai + (size_t)192 * 1024];
  }
  y2p[idx] = v;
}

// ---------------------------------------------------------------------------
// Final: y1 affine (on the fly), concat [y1, y2], actnorm, permutation gather.
// ---------------------------------------------------------------------------
__global__ __launch_bounds__(256)
void final_kernel(const float* __restrict__ x, const float* __restrict__ a2,
                  const float* __restrict__ y2p, const int* __restrict__ pidx,
                  const float* __restrict__ act_norm, const float* __restrict__ act_offset,
                  float* __restrict__ out) {
  const int idx = blockIdx.x * blockDim.x + threadIdx.x;
  if (idx >= 32 * 384 * 32 * 32) return;
  const int w = idx & 31;
  const int h = (idx >> 5) & 31;
  const int o = (idx >> 10) % 384;
  const int b = idx / (384 * 1024);

  const int c = pidx[o];
  const float scale = 0.2f * log1pf(expf(0.5f * act_norm[c]));

  float v;
  if (c >= 192) {
    v = y2p[(((size_t)b * 192 + (c - 192)) * 34 + h + 1) * 34 + w + 1];
  } else {
    const size_t ai = (((size_t)b * 384 + c) * 32 + h) * 32 + w;
    const float sj = 2.0f * tanhf(0.2f * a2[ai]);
    const float dn = x[(((size_t)b * 96 + (c >> 2)) * 64 + (2 * h + ((c >> 1) & 1))) * 64 +
                       (2 * w + (c & 1))];
    v = dn * expf(sj) + a2[ai + (size_t)192 * 1024];
  }
  out[idx] = v * scale + act_offset[c];
}

// ---------------------------------------------------------------------------
extern "C" void kernel_launch(void* const* d_in, const int* in_sizes, int n_in,
                              void* d_out, int out_size, void* d_ws, size_t ws_size,
                              hipStream_t stream) {
  const float* x          = (const float*)d_in[0];
  const float* w_hi       = (const float*)d_in[1];
  const float* b_hi       = (const float*)d_in[2];
  const float* w_lo       = (const float*)d_in[3];
  const float* b_lo       = (const float*)d_in[4];
  const float* act_norm   = (const float*)d_in[5];
  const float* act_offset = (const float*)d_in[6];
  const float* perm_w     = (const float*)d_in[7];
  float* out = (float*)d_out;

  // Workspace: A (a1, then aliased by a2) | R (d1p f16, then y2p f32) | wT | LUT
  const size_t A_ELEMS = (size_t)32 * 384 * 32 * 32;        // 12,582,912 f32
  const size_t P_ELEMS = (size_t)32 * 192 * 34 * 34;        //  7,102,464 padded elems
  float*    A    = (float*)d_ws;
  float*    R    = A + A_ELEMS;                             // 28,409,856 B region
  _Float16* d1p  = (_Float16*)R;                            // f16, dead after conv_hi
  float*    y2p  = R;                                       // f32, aliases d1p
  _Float16* whT  = (_Float16*)(R + P_ELEMS);                // 4*384*192 f16
  _Float16* wloT = whT + (size_t)4 * 384 * 192;             // 9*384*192 f16
  int*      pidx = (int*)(wloT + (size_t)9 * 384 * 192);

  // 0) prep: LUT + weight transposes + padded down(x1)
  perm_idx_kernel<<<2, 256, 0, stream>>>(perm_w, pidx);
  whi_remap_kernel<<<(4 * 384 * 192) / 256, 256, 0, stream>>>(w_hi, whT);
  wlo_transpose_kernel<<<(9 * 384 * 192) / 256, 256, 0, stream>>>(w_lo, wloT);
  down_pad_kernel<<<(int)(P_ELEMS / 256), 256, 0, stream>>>(x, d1p);

  // 1) a1 = conv_hi(x1) as 2x2 conv over down(x1)
  conv_wmma<2, _Float16>
      <<<dim3(3, 16, 32), 256, 0, stream>>>(d1p, whT, b_hi, A);

  // 2) y2 (padded) = down(x2) * exp(2 tanh(0.2 a1_lo)) + a1_hi
  affine_y2p_kernel<<<(int)(P_ELEMS / 256), 256, 0, stream>>>(x, A, y2p);

  // 3) a2 = conv_lo(y2), 3x3 (aliases a1's buffer)
  conv_wmma<3, float>
      <<<dim3(3, 16, 32), 256, 0, stream>>>(y2p, wloT, b_lo, A);

  // 4) y1 affine + concat + actnorm + permutation -> out
  final_kernel<<<(32 * 384 * 1024) / 256, 256, 0, stream>>>(
      x, A, y2p, pidx, act_norm, act_offset, out);
}
